// CNN_88287347736632
// MI455X (gfx1250) — matrix operand order repaired and checked by reading, not compiled, not run on hardware
//
#include <hip/hip_runtime.h>

typedef __attribute__((ext_vector_type(16))) _Float16 v16h;
typedef __attribute__((ext_vector_type(8)))  _Float16 v8h;
typedef __attribute__((ext_vector_type(8)))  float    v8f;

// ---------------------------------------------------------------------------
// Elementwise helpers
// ---------------------------------------------------------------------------
__global__ void bn_kernel(const float* __restrict__ b, const float* __restrict__ g,
                          const float* __restrict__ be, const float* __restrict__ m,
                          const float* __restrict__ v,
                          float* __restrict__ alpha, float* __restrict__ beta, int c) {
  int i = blockIdx.x * blockDim.x + threadIdx.x;
  if (i < c) {
    float s = g[i] * rsqrtf(v[i] + 1e-5f);
    alpha[i] = s;
    beta[i]  = (b[i] - m[i]) * s + be[i];
  }
}

// Weight repack: OIHW f32 -> [o][ (kh*KW+kw)*Cin + ci ] f16  (K-order: s outer, ci inner)
__global__ void wperm_kernel(const float* __restrict__ src, _Float16* __restrict__ dst,
                             int Cout, int Cin, int KH, int KW) {
  int Ktot = Cin * KH * KW;
  int i = blockIdx.x * blockDim.x + threadIdx.x;
  if (i >= Cout * Ktot) return;
  int o = i / Ktot;
  int r = i - o * Ktot;
  int s = r / Cin;
  int ci = r - s * Cin;
  int kh = s / KW;
  int kw = s - kh * KW;
  dst[i] = (_Float16)src[((o * Cin + ci) * KH + kh) * KW + kw];
}

// Window extraction: x (B,T,F) f32 -> in0 (chunk)(F)(WL) f16  (NHWC with C=1)
__global__ void win_kernel(const float* __restrict__ x, _Float16* __restrict__ in0,
                           int n0, int chunk, int T, int F, int ctx) {
  int wl = 2 * ctx + 1;
  int idx = blockIdx.x * blockDim.x + threadIdx.x;
  int total = chunk * F * wl;
  if (idx >= total) return;
  int j  = idx % wl;
  int f  = (idx / wl) % F;
  int ln = idx / (wl * F);
  int gn = n0 + ln;
  int b = gn / T, t = gn % T;
  int ts = t + j - ctx;
  float v = (ts >= 0 && ts < T) ? x[(b * T + ts) * F + f] : 0.0f;
  in0[idx] = (_Float16)v;
}

// conv1: Cin=1, 3x3 SAME, direct VALU. Output NHWC: out[(n*P + p)*32 + co].
__global__ void conv1_kernel(const _Float16* __restrict__ in, const float* __restrict__ W1,
                             const float* __restrict__ alpha, const float* __restrict__ beta,
                             _Float16* __restrict__ out, int CH) {
  const int H = 144, W = 15, P = H * W;
  int idx = blockIdx.x * blockDim.x + threadIdx.x;
  if (idx >= CH * P * 32) return;
  int co = idx % 32;
  int p  = (idx / 32) % P;
  int n  = idx / (32 * P);
  int w = p % W, h = p / W;
  const _Float16* src = in + (size_t)n * P;
  float acc = 0.0f;
#pragma unroll
  for (int kh = 0; kh < 3; ++kh)
#pragma unroll
    for (int kw = 0; kw < 3; ++kw) {
      int ih = h + kh - 1, iw = w + kw - 1;
      bool ok = ((unsigned)ih < (unsigned)H) && ((unsigned)iw < (unsigned)W);
      int si = ok ? (ih * W + iw) : 0;
      float xv = (float)src[si];
      acc += W1[co * 9 + kh * 3 + kw] * (ok ? xv : 0.0f);
    }
  out[idx] = (_Float16)fmaxf(acc * alpha[co] + beta[co], 0.0f);
}

// Max-pool (2,1) over H, NHWC: row = WC = W*C contiguous halves.
__global__ void pool2x1_kernel(const _Float16* __restrict__ in, _Float16* __restrict__ out,
                               int Nimg, int Hout, int WC) {
  int idx = blockIdx.x * blockDim.x + threadIdx.x;
  if (idx >= Nimg * Hout * WC) return;
  int wc = idx % WC;
  int h  = (idx / WC) % Hout;
  int n  = idx / (WC * Hout);
  const _Float16* p = in + ((size_t)n * (2 * Hout) + 2 * h) * WC + wc;
  out[idx] = (_Float16)fmaxf((float)p[0], (float)p[WC]);
}

// Spatial mean over P positions, NHWC input (n, P, C) -> feat (n, C) f32
__global__ void mean_kernel(const _Float16* __restrict__ in, float* __restrict__ feat,
                            int Nimg, int P, int C) {
  int idx = blockIdx.x * blockDim.x + threadIdx.x;
  if (idx >= Nimg * C) return;
  int c = idx % C;
  int n = idx / C;
  const _Float16* p = in + (size_t)n * P * C + c;
  float s = 0.0f;
  for (int i = 0; i < P; ++i) s += (float)p[(size_t)i * C];
  feat[idx] = s / (float)P;
}

__global__ void logits_kernel(const float* __restrict__ feat, const float* __restrict__ W8,
                              const float* __restrict__ b8, float* __restrict__ out,
                              int n0, int chunk, int C, int NC) {
  int idx = blockIdx.x * blockDim.x + threadIdx.x;
  if (idx >= chunk * NC) return;
  int c = idx % NC;
  int n = idx / NC;
  const float* f = feat + (size_t)n * C;
  const float* w = W8 + (size_t)c * C;
  float s = b8[c];
  for (int i = 0; i < C; ++i) s += f[i] * w[i];
  out[(size_t)(n0 + n) * NC + c] = s;
}

// ---------------------------------------------------------------------------
// Implicit-GEMM conv + BN + ReLU via V_WMMA_F32_16X16X32_F16 (wave32), NHWC.
// Requires: Cout%32==0, Cin%32==0, Ktot%32==0 (true for conv2..conv7).
//
// One wave computes a 32(Cout) x 64(positions) tile: 2 A fragments x 4 B
// fragments -> 8 accumulators. Per K-step: 4 b128 (A) + 8 b128 (B) loads and
// 8 v_wmma (1.5 loads/wmma). B fragments are 16 contiguous ci (NHWC); D
// stores are 8 contiguous channels -> aligned b128 stores.
//
// Fragment layouts used here (A/B K order contiguous, same in both; not the ISA table order):
//   A 16x32 f16 : lane row M = lane&15, half h = lane>>4;
//                 elem e -> K = k0 + 16h + e
//   B 32x16 f16 : lane col N = lane&15; elem e -> K = k0 + 16h + e
//   C/D 16x16 f32: elem v -> row M = v + 8h, col N = lane&15
// ---------------------------------------------------------------------------
template <bool PAD>
__global__ void __launch_bounds__(32)
conv_wmma(const _Float16* __restrict__ in,
          const _Float16* __restrict__ wt,
          const float* __restrict__ alpha,
          const float* __restrict__ beta,
          _Float16* __restrict__ out,
          int Cin, int Hin, int Win,
          int Cout, int Hout, int Wout,
          int KW, int ph, int pw, int Ktot) {
  const int P  = Hout * Wout;
  const int pT = (P + 63) >> 6;
  const int cW = Cout >> 5;            // 32-row c-tiles

  int bid = blockIdx.x;
  int pt = bid % pT;
  int t2 = bid / pT;
  int ct = t2 % cW;
  int n  = t2 / cW;

  int lane   = threadIdx.x;
  int laneM  = lane & 15;
  int halfId = lane >> 4;

  int ohh[4], oww[4], posOff[4];
  bool cval[4];
#pragma unroll
  for (int j = 0; j < 4; ++j) {
    int c = pt * 64 + j * 16 + laneM;
    cval[j] = (c < P);
    int cc = cval[j] ? c : (P - 1);
    ohh[j] = cc / Wout;
    oww[j] = cc - ohh[j] * Wout;
    posOff[j] = ohh[j] * Win + oww[j];     // top-left spatial index (VALID case)
  }

  v8f acc[2][4] = {};
  const _Float16* wrow0 = wt + (size_t)(ct * 32 + laneM) * Ktot;
  const _Float16* wrow1 = wrow0 + (size_t)16 * Ktot;
  const _Float16* inN   = in + (size_t)n * Hin * Win * Cin;
  const int hOfs = halfId * 16;            // half-wave ci offset within fragment

  for (int k0 = 0; k0 < Ktot; k0 += 32) {
    // ---- A fragments (rows ct*32 + [0,16) and + [16,32)) ----
    v8h a0lo = *(const v8h*)(wrow0 + k0 + halfId * 16);
    v8h a0hi = *(const v8h*)(wrow0 + k0 + 8 + halfId * 16);
    v8h a1lo = *(const v8h*)(wrow1 + k0 + halfId * 16);
    v8h a1hi = *(const v8h*)(wrow1 + k0 + 8 + halfId * 16);
    v16h a0 = __builtin_shufflevector(a0lo, a0hi,
                                      0, 1, 2, 3, 4, 5, 6, 7,
                                      8, 9, 10, 11, 12, 13, 14, 15);
    v16h a1 = __builtin_shufflevector(a1lo, a1hi,
                                      0, 1, 2, 3, 4, 5, 6, 7,
                                      8, 9, 10, 11, 12, 13, 14, 15);

    // ---- per-step uniform spatial position (same kh,kw for whole step) ----
    int s  = k0 / Cin;
    int ciBase = k0 - s * Cin + hOfs;
    int kh = s / KW;
    int kw = s - kh * KW;

    // ---- B fragments: 16 contiguous ci (NHWC) -> two b128 loads each ----
    v16h bfr[4];
    bool msk[4];
#pragma unroll
    for (int j = 0; j < 4; ++j) {
      int so;
      if (PAD) {
        int ih = ohh[j] + kh - ph;
        int iw = oww[j] + kw - pw;
        msk[j] = ((unsigned)ih < (unsigned)Hin) && ((unsigned)iw < (unsigned)Win);
        so = msk[j] ? (ih * Win + iw) : 0;
      } else {
        so = posOff[j] + kh * Win + kw;
      }
      const _Float16* bp = inN + (size_t)so * Cin + ciBase;
      v8h blo = *(const v8h*)(bp);
      v8h bhi = *(const v8h*)(bp + 8);
      bfr[j] = __builtin_shufflevector(blo, bhi,
                                       0, 1, 2, 3, 4, 5, 6, 7,
                                       8, 9, 10, 11, 12, 13, 14, 15);
    }
    if (PAD) {
      v16h zf = {};
#pragma unroll
      for (int j = 0; j < 4; ++j) bfr[j] = msk[j] ? bfr[j] : zf;
    }

#pragma unroll
    for (int j = 0; j < 4; ++j) {
      acc[0][j] = __builtin_amdgcn_wmma_f32_16x16x32_f16(
                      false, a0, false, bfr[j], (short)0, acc[0][j], false, false);
      acc[1][j] = __builtin_amdgcn_wmma_f32_16x16x32_f16(
                      false, a1, false, bfr[j], (short)0, acc[1][j], false, false);
    }
  }

  // ---- Epilogue: fused BN + ReLU; 8 consecutive channels -> 1 b128 store ----
#pragma unroll
  for (int gr = 0; gr < 2; ++gr) {
    int rbase = ct * 32 + gr * 16 + halfId * 8;
    const float* ap = alpha + rbase;
    const float* bp = beta  + rbase;
    float al[8], bt[8];
#pragma unroll
    for (int v = 0; v < 8; ++v) { al[v] = ap[v]; bt[v] = bp[v]; }
#pragma unroll
    for (int j = 0; j < 4; ++j) {
      if (!cval[j]) continue;
      int c = pt * 64 + j * 16 + laneM;
      v8h ov;
#pragma unroll
      for (int v = 0; v < 8; ++v)
        ov[v] = (_Float16)fmaxf(acc[gr][j][v] * al[v] + bt[v], 0.0f);
      *(v8h*)(out + ((size_t)n * P + c) * Cout + rbase) = ov;
    }
  }
}

// ---------------------------------------------------------------------------
// Host launcher
// ---------------------------------------------------------------------------
extern "C" void kernel_launch(void* const* d_in, const int* in_sizes, int n_in,
                              void* d_out, int out_size, void* d_ws, size_t ws_size,
                              hipStream_t stream) {
  (void)in_sizes; (void)n_in; (void)out_size; (void)ws_size;
  const int B = 16, T = 108, F = 144, CTX = 7, WL = 15, NC = 25;
  const int N = B * T;        // 1728
  const int CH = 216;         // ping-pong chunk buffers (~30MB) stay L2-resident

  const float* x = (const float*)d_in[0];
  const float *W[8], *bb[8], *g[8], *be[8], *m[8], *vv[8];
  for (int l = 1; l <= 7; ++l) {
    int base = 2 + (l - 1) * 6;
    W[l]  = (const float*)d_in[base + 0];
    bb[l] = (const float*)d_in[base + 1];
    g[l]  = (const float*)d_in[base + 2];
    be[l] = (const float*)d_in[base + 3];
    m[l]  = (const float*)d_in[base + 4];
    vv[l] = (const float*)d_in[base + 5];
  }
  const float* W8 = (const float*)d_in[44];
  const float* b8 = (const float*)d_in[45];

  struct LC { int Cin, Hin, Win, Cout, Hout, Wout, KH, KW, ph, pw; };
  const LC L[8] = {
    {0,0,0,0,0,0,0,0,0,0},
    {1, 144, 15, 32, 144, 15, 3, 3, 1, 1},
    {32,144, 15, 32, 144, 15, 3, 3, 1, 1},
    {32,144, 15, 32, 144, 15, 3, 3, 1, 1},
    {32,144, 15, 32, 144, 15, 3, 3, 1, 1},
    {32, 72, 15, 64, 70, 13, 3, 3, 0, 0},
    {64, 70, 13, 64, 68, 11, 3, 3, 0, 0},
    {64, 34, 11, 128, 23, 3, 12, 9, 0, 0},
  };

  size_t off = 0;
  auto alloc = [&](size_t bytes) {
    size_t o = off; off += (bytes + 255) & ~(size_t)255; return o;
  };
  char* ws = (char*)d_ws;
  const size_t maxAct = (size_t)CH * 32 * 144 * 15;
  _Float16* bufA = (_Float16*)(ws + alloc(maxAct * sizeof(_Float16)));
  _Float16* bufB = (_Float16*)(ws + alloc(maxAct * sizeof(_Float16)));
  _Float16* wt16[8]; float* alp[8]; float* bet[8];
  for (int l = 2; l <= 7; ++l) {
    size_t we = (size_t)L[l].Cout * L[l].Cin * L[l].KH * L[l].KW;
    wt16[l] = (_Float16*)(ws + alloc(we * sizeof(_Float16)));
  }
  for (int l = 1; l <= 7; ++l) {
    alp[l] = (float*)(ws + alloc((size_t)L[l].Cout * sizeof(float)));
    bet[l] = (float*)(ws + alloc((size_t)L[l].Cout * sizeof(float)));
  }
  float* feat = (float*)(ws + alloc((size_t)CH * 128 * sizeof(float)));

  // --- weight repack + folded BN params ---
  for (int l = 2; l <= 7; ++l) {
    int we = L[l].Cout * L[l].Cin * L[l].KH * L[l].KW;
    wperm_kernel<<<(we + 255) / 256, 256, 0, stream>>>(W[l], wt16[l],
                                                       L[l].Cout, L[l].Cin, L[l].KH, L[l].KW);
  }
  for (int l = 1; l <= 7; ++l)
    bn_kernel<<<(L[l].Cout + 63) / 64, 64, 0, stream>>>(bb[l], g[l], be[l], m[l], vv[l],
                                                        alp[l], bet[l], L[l].Cout);

  for (int n0 = 0; n0 < N; n0 += CH) {
    _Float16* cur = bufB;
    _Float16* nxt = bufA;

    { // window extraction -> bufB (NHWC, C=1)
      int total = CH * F * WL;
      win_kernel<<<(total + 255) / 256, 256, 0, stream>>>(x, cur, n0, CH, T, F, CTX);
    }
    { // conv1 (direct) -> NHWC
      int total = CH * 144 * 15 * 32;
      conv1_kernel<<<(total + 255) / 256, 256, 0, stream>>>(cur, W[1], alp[1], bet[1], nxt, CH);
      _Float16* t = cur; cur = nxt; nxt = t;
    }

    auto runSame = [&](int l) {
      const LC& c = L[l];
      int P = c.Hout * c.Wout, pT = (P + 63) >> 6, cW = c.Cout >> 5;
      conv_wmma<true><<<(unsigned)(CH * cW * pT), 32, 0, stream>>>(
          cur, wt16[l], alp[l], bet[l], nxt,
          c.Cin, c.Hin, c.Win, c.Cout, c.Hout, c.Wout,
          c.KW, c.ph, c.pw, c.Cin * c.KH * c.KW);
      _Float16* t = cur; cur = nxt; nxt = t;
    };
    auto runValid = [&](int l) {
      const LC& c = L[l];
      int P = c.Hout * c.Wout, pT = (P + 63) >> 6, cW = c.Cout >> 5;
      conv_wmma<false><<<(unsigned)(CH * cW * pT), 32, 0, stream>>>(
          cur, wt16[l], alp[l], bet[l], nxt,
          c.Cin, c.Hin, c.Win, c.Cout, c.Hout, c.Wout,
          c.KW, c.ph, c.pw, c.Cin * c.KH * c.KW);
      _Float16* t = cur; cur = nxt; nxt = t;
    };

    runSame(2); runSame(3); runSame(4);
    { int total = CH * 72 * 15 * 32;   // pool1 NHWC: WC = 15*32
      pool2x1_kernel<<<(total + 255) / 256, 256, 0, stream>>>(cur, nxt, CH, 72, 15 * 32);
      _Float16* t = cur; cur = nxt; nxt = t; }
    runValid(5); runValid(6);
    { int total = CH * 34 * 11 * 64;   // pool2 NHWC: WC = 11*64
      pool2x1_kernel<<<(total + 255) / 256, 256, 0, stream>>>(cur, nxt, CH, 34, 11 * 64);
      _Float16* t = cur; cur = nxt; nxt = t; }
    runValid(7);
    { int total = CH * 128;            // mean over 69 positions
      mean_kernel<<<(total + 255) / 256, 256, 0, stream>>>(cur, feat, CH, 69, 128); }
    logits_kernel<<<(CH * NC + 255) / 256, 256, 0, stream>>>(feat, W8, b8, (float*)d_out,
                                                             n0, CH, 128, NC);
  }
}
